// MultiHeadAtt_69518340653730
// MI455X (gfx1250) — matrix operand order repaired and checked by reading, not compiled, not run on hardware
//
#include <hip/hip_runtime.h>
#include <hip/hip_fp16.h>

// ---------------------------------------------------------------------------
// MI455X / gfx1250 multi-head attention, all matmuls via v_wmma_f32_16x16x32_f16
// B=2, S=2048, D=1024, H=16, e=64
// ---------------------------------------------------------------------------

typedef _Float16 v16h __attribute__((ext_vector_type(16)));
typedef float    v8f  __attribute__((ext_vector_type(8)));

#define BB 2
#define SS 2048
#define DD 1024
#define HH 16
#define EH 64
#define MM (BB * SS)          // 4096 rows for all projection GEMMs

union Frag {                  // 8 dwords = 16 f16 = one WMMA A/B operand
    v16h h;
    unsigned int u[8];
};

static __device__ inline v8f wmma_f16(const Frag& a, const Frag& b, v8f c) {
    return __builtin_amdgcn_wmma_f32_16x16x32_f16(
        /*neg_a=*/false, a.h, /*neg_b=*/false, b.h,
        /*c_mod=*/(short)0, c, /*reuse_a=*/false, /*reuse_b=*/false);
}

// ---------------------------------------------------------------------------
// fp32 -> f16 conversion
// ---------------------------------------------------------------------------
__global__ void cvt_f32_f16(const float* __restrict__ src,
                            _Float16* __restrict__ dst, int n) {
    int i = blockIdx.x * blockDim.x + threadIdx.x;
    if (i < n) dst[i] = (_Float16)src[i];
}

// ---------------------------------------------------------------------------
// C[M,N] = A[M,K] (f16, row-major) x W[N,K]^T (f16, row-major).
// One wave computes a 32x64 tile: 2 (M) x 4 (N) accumulators.
// Per K-step (32): 2 A frags + 4 B frags -> 8 WMMAs (1.5 b128 loads / wmma).
// Epilogue modes:
//   0: f16 out, row-major [M,N]
//   1: f16 out, per-head layout  out[((h*BB+b)*SS+s)*EH + e]   (q/k)
//   2: f16 out, per-head transposed out[((h*BB+b)*EH+e)*SS + s] (vT)
//   3: f32 out, row-major [M,N] (final projection)
// ---------------------------------------------------------------------------
__global__ __launch_bounds__(256) void gemm_wmma(
    const _Float16* __restrict__ A, const _Float16* __restrict__ W,
    void* __restrict__ out, int M, int N, int K, int mode) {
    const int lane = threadIdx.x & 31;
    const int wave = threadIdx.x >> 5;
    const int half = lane >> 4;
    const int l16  = lane & 15;

    const int tile = blockIdx.x * 8 + wave;     // 32x64 macro tiles
    const int ntn  = N >> 6;                    // N / 64
    const int tm   = tile / ntn;
    const int tn   = tile - tm * ntn;
    if (tm >= (M >> 5)) return;
    const int m0 = tm << 5;
    const int n0 = tn << 6;

    const v8f vzero = {0.f, 0.f, 0.f, 0.f, 0.f, 0.f, 0.f, 0.f};
    v8f acc[2][4];
#pragma unroll
    for (int mi = 0; mi < 2; ++mi)
#pragma unroll
        for (int ni = 0; ni < 4; ++ni) acc[mi][ni] = vzero;

    const _Float16* arow0 = A + (size_t)(m0 + l16) * K;        // M rows 0..15
    const _Float16* arow1 = A + (size_t)(m0 + 16 + l16) * K;   // M rows 16..31
    const _Float16* brow0 = W + (size_t)(n0 + l16) * K;        // N cols, W rows
    const size_t bstride  = (size_t)16 * K;

    for (int k0 = 0; k0 < K; k0 += 32) {
        Frag a[2], b[4];
#pragma unroll
        for (int i = 0; i < 8; ++i) {
            // A 16x32 f16 layout: lane holds 16 contiguous K at k0 + half*16
            const int kk = k0 + (half << 4) + (i << 1);
            a[0].u[i] = *(const unsigned int*)(arow0 + kk);
            a[1].u[i] = *(const unsigned int*)(arow1 + kk);
        }
#pragma unroll
        for (int ni = 0; ni < 4; ++ni) {
            // B 32x16 f16 layout: lane holds 16 contiguous K at k0 + half*16
            b[ni].h = *(const v16h*)(brow0 + (size_t)ni * bstride + k0 +
                                     (half << 4));
        }
#pragma unroll
        for (int mi = 0; mi < 2; ++mi)
#pragma unroll
            for (int ni = 0; ni < 4; ++ni)
                acc[mi][ni] = wmma_f16(a[mi], b[ni], acc[mi][ni]);
    }

    // D layout: VGPR j -> row (msub + j + 8*half), col (nsub + l16)
#pragma unroll
    for (int mi = 0; mi < 2; ++mi) {
#pragma unroll
        for (int ni = 0; ni < 4; ++ni) {
#pragma unroll
            for (int j = 0; j < 8; ++j) {
                const int m = m0 + (mi << 4) + j + (half << 3);
                const int n = n0 + (ni << 4) + l16;
                const float v = acc[mi][ni][j];
                if (mode == 0) {
                    ((_Float16*)out)[(size_t)m * N + n] = (_Float16)v;
                } else if (mode == 1) {
                    const int h = n >> 6, e = n & 63;
                    const int b_ = m >> 11, s = m & (SS - 1);
                    ((_Float16*)out)[(((size_t)(h * BB + b_) * SS + s) << 6) +
                                     e] = (_Float16)v;
                } else if (mode == 2) {
                    const int h = n >> 6, e = n & 63;
                    const int b_ = m >> 11, s = m & (SS - 1);
                    ((_Float16*)out)[((size_t)(h * BB + b_) * EH + e) * SS +
                                     s] = (_Float16)v;
                } else {
                    ((float*)out)[(size_t)m * N + n] = v;
                }
            }
        }
    }
}

// ---------------------------------------------------------------------------
// Flash attention: one wave per (hb, 16-row block).
//   q,k : [h*BB+b][s][e] f16,  vT : [h*BB+b][e][s] f16
//   out : [b*SS+s][h*EH+e] f16 (concatenated heads, ready for WO projection)
// ---------------------------------------------------------------------------
__global__ __launch_bounds__(32) void flash_attn(
    const _Float16* __restrict__ qh, const _Float16* __restrict__ kh,
    const _Float16* __restrict__ vT, _Float16* __restrict__ attn_out) {
    __shared__ _Float16 lds[16 * 32];   // P tile re-swizzle buffer (1 KB)

    const int lane = threadIdx.x & 31;
    const int half = lane >> 4;
    const int l16  = lane & 15;
    const int hb   = blockIdx.y;        // h*BB + b
    const int h    = hb >> 1;
    const int b    = hb & 1;
    const int row0 = blockIdx.x << 4;

    const _Float16* q  = qh + (size_t)hb * SS * EH;
    const _Float16* k  = kh + (size_t)hb * SS * EH;
    const _Float16* vt = vT + (size_t)hb * EH * SS;

    // Q fragments for this 16-row block: K = 64 -> two A fragments
    Frag aq0, aq1;
    const _Float16* qrow = q + (size_t)(row0 + l16) * EH;
#pragma unroll
    for (int i = 0; i < 8; ++i) {
        const int kk = (half << 4) + (i << 1);
        aq0.u[i] = *(const unsigned int*)(qrow + kk);
        aq1.u[i] = *(const unsigned int*)(qrow + 32 + kk);
    }

    const v8f vzero = {0.f, 0.f, 0.f, 0.f, 0.f, 0.f, 0.f, 0.f};
    v8f oacc[4];
    oacc[0] = vzero; oacc[1] = vzero; oacc[2] = vzero; oacc[3] = vzero;
    float Mrun[8], Lrun[8];
#pragma unroll
    for (int j = 0; j < 8; ++j) { Mrun[j] = -1e30f; Lrun[j] = 0.f; }

    const float scale = 0.125f;         // 1/sqrt(64)

    for (int t0 = 0; t0 < SS; t0 += 32) {
        // ---- scores tile 16x32 = two 16x16 accumulators (K = e = 64) ----
        v8f s0 = vzero, s1 = vzero;
        {
            Frag b0, b1;
            b0.h = *(const v16h*)(k + (size_t)(t0 + l16) * EH + (half << 4));
            b1.h = *(const v16h*)(k + (size_t)(t0 + l16) * EH + 32 + (half << 4));
            s0 = wmma_f16(aq0, b0, s0);
            s0 = wmma_f16(aq1, b1, s0);
            Frag c0, c1;
            c0.h = *(const v16h*)(k + (size_t)(t0 + 16 + l16) * EH + (half << 4));
            c1.h = *(const v16h*)(k + (size_t)(t0 + 16 + l16) * EH + 32 + (half << 4));
            s1 = wmma_f16(aq0, c0, s1);
            s1 = wmma_f16(aq1, c1, s1);
        }

        // ---- online softmax (row stats per D-VGPR j; half-wave reduce) ----
        float p0[8], p1[8];
#pragma unroll
        for (int j = 0; j < 8; ++j) {
            float x0 = s0[j] * scale;
            float x1 = s1[j] * scale;
            float mx = fmaxf(x0, x1);
            for (int msk = 1; msk < 16; msk <<= 1)
                mx = fmaxf(mx, __shfl_xor(mx, msk, 32));
            const float Mn    = fmaxf(Mrun[j], mx);
            const float alpha = __expf(Mrun[j] - Mn);
            Mrun[j] = Mn;
            p0[j] = __expf(x0 - Mn);
            p1[j] = __expf(x1 - Mn);
            float ts = p0[j] + p1[j];
            for (int msk = 1; msk < 16; msk <<= 1)
                ts += __shfl_xor(ts, msk, 32);
            Lrun[j] = Lrun[j] * alpha + ts;
            oacc[0][j] *= alpha; oacc[1][j] *= alpha;
            oacc[2][j] *= alpha; oacc[3][j] *= alpha;
        }

        // ---- re-swizzle P (D layout) -> A fragment via LDS ----
        __syncthreads();
#pragma unroll
        for (int j = 0; j < 8; ++j) {
            const int r = j + (half << 3);
            lds[r * 32 + l16]      = (_Float16)p0[j];
            lds[r * 32 + 16 + l16] = (_Float16)p1[j];
        }
        __syncthreads();
        Frag pa;
#pragma unroll
        for (int i = 0; i < 8; ++i) {
            const int kk = (half << 4) + (i << 1);
            pa.u[i] = *(const unsigned int*)&lds[l16 * 32 + kk];
        }

        // ---- O += P(16x32) @ V(32x64): 4 N-tiles, B cols = vT rows ----
#pragma unroll
        for (int et = 0; et < 4; ++et) {
            Frag bv;
            bv.h = *(const v16h*)(vt + (size_t)(et * 16 + l16) * SS + t0 +
                                  (half << 4));
            oacc[et] = wmma_f16(pa, bv, oacc[et]);
        }
    }

    // ---- normalize and emit concatenated-head f16 output ----
#pragma unroll
    for (int j = 0; j < 8; ++j) {
        const float inv = 1.0f / Lrun[j];
        const int s   = row0 + j + (half << 3);
        const size_t row = (size_t)b * SS + s;
#pragma unroll
        for (int et = 0; et < 4; ++et) {
            const int col = (h << 6) + (et << 4) + l16;
            attn_out[row * DD + col] = (_Float16)(oacc[et][j] * inv);
        }
    }
}

// ---------------------------------------------------------------------------
// Host-side launch
// ---------------------------------------------------------------------------
extern "C" void kernel_launch(void* const* d_in, const int* in_sizes, int n_in,
                              void* d_out, int out_size, void* d_ws,
                              size_t ws_size, hipStream_t stream) {
    (void)in_sizes; (void)n_in; (void)out_size; (void)ws_size;

    const float* E  = (const float*)d_in[0];
    const float* WQ = (const float*)d_in[1];
    const float* WK = (const float*)d_in[2];
    const float* WV = (const float*)d_in[3];
    const float* WO = (const float*)d_in[4];
    const float* HQ = (const float*)d_in[5];
    const float* HK = (const float*)d_in[6];
    const float* HV = (const float*)d_in[7];

    const size_t NE = (size_t)MM * DD;   // 4,194,304 (E / Q / K / V / heads / o)
    const size_t NW = (size_t)DD * DD;   // 1,048,576 (each weight)

    _Float16* ws = (_Float16*)d_ws;      // ~82 MB of f16 scratch total
    _Float16* E16   = ws;                ws += NE;
    _Float16* WQ16  = ws;                ws += NW;
    _Float16* WK16  = ws;                ws += NW;
    _Float16* WV16  = ws;                ws += NW;
    _Float16* WO16  = ws;                ws += NW;
    _Float16* HQ16  = ws;                ws += NW;
    _Float16* HK16  = ws;                ws += NW;
    _Float16* HV16  = ws;                ws += NW;
    _Float16* Q16   = ws;                ws += NE;
    _Float16* K16   = ws;                ws += NE;
    _Float16* V16   = ws;                ws += NE;
    _Float16* qhd   = ws;                ws += NE;
    _Float16* khd   = ws;                ws += NE;
    _Float16* vTd   = ws;                ws += NE;
    _Float16* att16 = ws;                ws += NE;

    // 1) fp32 -> f16
    cvt_f32_f16<<<(int)((NE + 255) / 256), 256, 0, stream>>>(E, E16, (int)NE);
    cvt_f32_f16<<<(int)((NW + 255) / 256), 256, 0, stream>>>(WQ, WQ16, (int)NW);
    cvt_f32_f16<<<(int)((NW + 255) / 256), 256, 0, stream>>>(WK, WK16, (int)NW);
    cvt_f32_f16<<<(int)((NW + 255) / 256), 256, 0, stream>>>(WV, WV16, (int)NW);
    cvt_f32_f16<<<(int)((NW + 255) / 256), 256, 0, stream>>>(WO, WO16, (int)NW);
    cvt_f32_f16<<<(int)((NW + 255) / 256), 256, 0, stream>>>(HQ, HQ16, (int)NW);
    cvt_f32_f16<<<(int)((NW + 255) / 256), 256, 0, stream>>>(HK, HK16, (int)NW);
    cvt_f32_f16<<<(int)((NW + 255) / 256), 256, 0, stream>>>(HV, HV16, (int)NW);

    // 32x64 wave tiles: (4096/32)*(1024/64) = 2048 tiles; 8 waves/block
    const int gblocks = (MM / 32) * (DD / 64) / 8;

    // 2) Q/K/V = E @ W^T
    gemm_wmma<<<gblocks, 256, 0, stream>>>(E16, WQ16, Q16, MM, DD, DD, 0);
    gemm_wmma<<<gblocks, 256, 0, stream>>>(E16, WK16, K16, MM, DD, DD, 0);
    gemm_wmma<<<gblocks, 256, 0, stream>>>(E16, WV16, V16, MM, DD, DD, 0);

    // 3) head projections: q/k in [h][b][s][e], v transposed [h][b][e][s]
    gemm_wmma<<<gblocks, 256, 0, stream>>>(Q16, HQ16, qhd, MM, DD, DD, 1);
    gemm_wmma<<<gblocks, 256, 0, stream>>>(K16, HK16, khd, MM, DD, DD, 1);
    gemm_wmma<<<gblocks, 256, 0, stream>>>(V16, HV16, vTd, MM, DD, DD, 2);

    // 4) flash attention: grid = (S/16 row blocks, H*B)
    flash_attn<<<dim3(SS / 16, HH * BB), 32, 0, stream>>>(qhd, khd, vTd, att16);

    // 5) final projection -> fp32 output
    gemm_wmma<<<gblocks, 256, 0, stream>>>(att16, WO16, d_out, MM, DD, DD, 3);
}
